// MultiheadAttentionMechanism_19000935318259
// MI455X (gfx1250) — matrix operand order repaired and checked by reading, not compiled, not run on hardware
//
#include <hip/hip_runtime.h>

typedef unsigned short u16;
typedef __attribute__((ext_vector_type(16))) __bf16 v16bf;
typedef __attribute__((ext_vector_type(8)))  float   v8f;

#define BSZ   2
#define SKLEN 2048
#define SQLEN 2048
#define SKDIM 512
#define SADIM 512
#define NH    8
#define DK    64

#define NEG_INF (-3.4028234663852886e+38f)

union Frag16 {
  v16bf bf;
  u16   u[16];
  uint4 q[2];
};

__device__ __forceinline__ u16 f32_bf16(float f) {
  unsigned int x = __float_as_uint(f);
  unsigned int r = 0x7fffu + ((x >> 16) & 1u);
  return (u16)((x + r) >> 16);
}

// ---- WMMA fragment loaders (wave32 layouts per CDNA5 ISA 7.12.2) ----

// A matrix 16x32 bf16, source row-major. lane L: row = row0 + (L%16);
// elements 0..7 hold K = kb..kb+7, 8..15 hold K = kb+16..kb+23, kb = k0 + 8*(L>=16).
__device__ __forceinline__ v16bf load_frag_a(const u16* __restrict__ A, int lda,
                                             int row0, int k0, int lane) {
  int r  = row0 + (lane & 15);
  int kb = k0 + ((lane >> 4) << 3);
  const u16* p = A + (size_t)r * lda + kb;
  Frag16 f;
  f.q[0] = *(const uint4*)(p);
  f.q[1] = *(const uint4*)(p + 16);
  return f.bf;
}

// A fragment from a wave-private LDS region laid out [16 rows][128 cols] bf16.
__device__ __forceinline__ v16bf load_frag_a_lds(const u16* p0, int k0, int lane) {
  const u16* p = p0 + (lane & 15) * 128 + k0 + ((lane >> 4) << 3);
  Frag16 f;
  f.q[0] = *(const uint4*)(p);
  f.q[1] = *(const uint4*)(p + 16);
  return f.bf;
}

// B matrix 32x16 bf16 from B^T stored row-major [N][K] with leading dim ldb.
// lane L: col n = n0 + (L%16); elements 0..7 hold K = kb..kb+7, 8..15 hold K = kb+16..kb+23, kb = k0 + 8*(L>=16).
__device__ __forceinline__ v16bf load_frag_b(const u16* __restrict__ Bt, int ldb,
                                             int n0, int k0, int lane) {
  int n  = n0 + (lane & 15);
  int kb = k0 + ((lane >> 4) << 3);
  const u16* p = Bt + (size_t)n * ldb + kb;
  Frag16 f;
  f.q[0] = *(const uint4*)(p);
  f.q[1] = *(const uint4*)(p + 16);
  return f.bf;
}

// ---------------- elementwise convert f32 -> bf16 ----------------

__global__ void __launch_bounds__(256)
f32_to_bf16_vec(const float* __restrict__ src, u16* __restrict__ dst, int n4) {
  int i = blockIdx.x * blockDim.x + threadIdx.x;
  if (i < n4) {
    float4 f = ((const float4*)src)[i];
    ushort4 h;
    h.x = f32_bf16(f.x); h.y = f32_bf16(f.y);
    h.z = f32_bf16(f.z); h.w = f32_bf16(f.w);
    ((ushort4*)dst)[i] = h;
  }
}

// ------------- 512x512 weight transpose + convert: Wt[n][k] = bf16(W[k][n]) -------------

__global__ void __launch_bounds__(256)
transpose512_to_bf16(const float* __restrict__ W, u16* __restrict__ Wt) {
  __shared__ float tile[32][33];
  int n0 = blockIdx.x * 32;
  int k0 = blockIdx.y * 32;
  int tx = threadIdx.x, ty = threadIdx.y;  // block (32,8)
#pragma unroll
  for (int j = 0; j < 32; j += 8)
    tile[ty + j][tx] = W[(size_t)(k0 + ty + j) * 512 + (n0 + tx)];
  __syncthreads();
#pragma unroll
  for (int j = 0; j < 32; j += 8)
    Wt[(size_t)(n0 + ty + j) * 512 + (k0 + tx)] = f32_bf16(tile[tx][ty + j]);
}

// ------------- per-head V transpose: Vt[b*H+h][d][k] = Vbf[b][k][h*DK+d] -------------

__global__ void __launch_bounds__(256)
transpose_v_heads(const u16* __restrict__ Vbf, u16* __restrict__ Vt) {
  __shared__ u16 tile[32][33];
  int bh = blockIdx.z;
  int b = bh >> 3, h = bh & 7;
  int k0 = blockIdx.x * 32;
  int d0 = blockIdx.y * 32;
  const u16* src = Vbf + (size_t)b * SKLEN * SADIM + h * DK;
  u16* dst = Vt + (size_t)bh * DK * SKLEN;
  int tx = threadIdx.x, ty = threadIdx.y;  // block (32,8)
#pragma unroll
  for (int j = 0; j < 32; j += 8)
    tile[ty + j][tx] = src[(size_t)(k0 + ty + j) * SADIM + (d0 + tx)];
  __syncthreads();
#pragma unroll
  for (int j = 0; j < 32; j += 8)
    dst[(size_t)(d0 + ty + j) * SKLEN + (k0 + tx)] = tile[tx][ty + j];
}

// ------------- GEMM: C[M][N] = A[M][K] * Bt[N][K]^T + bias -------------
// block = 256 threads = 8 waves, block tile 128x128, wave tile 64x32 (4x2 WMMA tiles)
// MODE 0: bf16 output, MODE 1: f32 output

template <int MODE>
__global__ void __launch_bounds__(256)
gemm_bf16_tn(const u16* __restrict__ A, const u16* __restrict__ Bt,
             const float* __restrict__ bias,
             u16* __restrict__ Cbf, float* __restrict__ Cf,
             int M, int N, int Kd) {
  const int lane = threadIdx.x & 31;
  const int wave = threadIdx.x >> 5;
  const int wm = wave >> 2;                 // 0..1
  const int wn = wave & 3;                  // 0..3
  const int m0 = blockIdx.x * 128 + wm * 64;
  const int n0 = blockIdx.y * 128 + wn * 32;

  v8f acc[4][2] = {};

  for (int kk = 0; kk < Kd; kk += 32) {
    v16bf a[4], b[2];
#pragma unroll
    for (int i = 0; i < 4; ++i) a[i] = load_frag_a(A, Kd, m0 + 16 * i, kk, lane);
#pragma unroll
    for (int j = 0; j < 2; ++j) b[j] = load_frag_b(Bt, Kd, n0 + 16 * j, kk, lane);
#pragma unroll
    for (int i = 0; i < 4; ++i)
#pragma unroll
      for (int j = 0; j < 2; ++j)
        acc[i][j] = __builtin_amdgcn_wmma_f32_16x16x32_bf16(
            false, a[i], false, b[j], (short)0, acc[i][j], false, false);
  }

  const int nlane = lane & 15;
  const int mhalf = (lane >> 4) << 3;
#pragma unroll
  for (int i = 0; i < 4; ++i) {
#pragma unroll
    for (int j = 0; j < 2; ++j) {
      int n = n0 + 16 * j + nlane;
      float bb = bias[n];
#pragma unroll
      for (int r = 0; r < 8; ++r) {
        int m = m0 + 16 * i + mhalf + r;
        float v = acc[i][j][r] + bb;
        if (MODE == 0) Cbf[(size_t)m * N + n] = f32_bf16(v);
        else           Cf[(size_t)m * N + n] = v;
      }
    }
  }
}

// ------------- fully fused attention: scores + mask + softmax + P@V -------------
// Per (b,h): block of 512 threads (16 waves) owns 16 q-rows x full 2048 k-cols.
// Wave w computes the 16x128 score strip via WMMA and keeps it in registers.
// Softmax reductions: lane -> 16-lane half group (shfl) -> waves (1KB of LDS).
// Normalized probabilities are written to aw (f32 output) and simultaneously
// transposed wave-locally through LDS into A-fragment layout to feed the P@V
// WMMAs; per-wave 16x64 partials are then block-reduced in LDS -> bf16 Ctx.
// Raw e and P never touch global memory.

__global__ void __launch_bounds__(512)
attn_fused(const u16* __restrict__ Qbf, const u16* __restrict__ Kbf,
           const int* __restrict__ mask, float* __restrict__ aw,
           const u16* __restrict__ Vt, u16* __restrict__ Ctx) {
  // 64KB multi-phase LDS: [softmax scratch] -> [P shuffle (u16)] -> [O partials]
  __shared__ __align__(16) float smem[16384];
  float (*part)[16] = (float(*)[16])smem;   // [wave][row], 1KB
  float* rowv = smem + 256;                 // [16]

  const int tid = threadIdx.x;
  const int lane = tid & 31;
  const int wave = tid >> 5;            // 0..15
  const int bh = blockIdx.y, b = bh >> 3, h = bh & 7;
  const int m0 = blockIdx.x * 16;       // q rows
  const int n0 = wave * 128;            // k col strip
  const u16* Qb = Qbf + (size_t)b * SQLEN * SADIM + h * DK;
  const u16* Kb = Kbf + (size_t)b * SKLEN * SADIM + h * DK;
  float* e = aw + (size_t)bh * SQLEN * SKLEN;
  const int* mk = mask + (size_t)b * SQLEN * SKLEN;

  v8f acc[8] = {};
#pragma unroll
  for (int kk = 0; kk < DK; kk += 32) {
    v16bf a = load_frag_a(Qb, SADIM, m0, kk, lane);
#pragma unroll
    for (int j = 0; j < 8; ++j) {
      v16bf bf = load_frag_b(Kb, SADIM, n0 + 16 * j, kk, lane);
      acc[j] = __builtin_amdgcn_wmma_f32_16x16x32_bf16(
          false, a, false, bf, (short)0, acc[j], false, false);
    }
  }

  const int nlane = lane & 15;
  const int mhalf = (lane >> 4) << 3;

  // scale + mask, in place
#pragma unroll
  for (int j = 0; j < 8; ++j) {
    int n = n0 + 16 * j + nlane;
#pragma unroll
    for (int r = 0; r < 8; ++r) {
      int m = m0 + mhalf + r;
      float v = acc[j][r] * 0.125f;  // 1/sqrt(64)
      if (mk[(size_t)m * SKLEN + n] == 0) v = NEG_INF;
      acc[j][r] = v;
    }
  }

  // ---- row max ----
  float mx[8];
#pragma unroll
  for (int r = 0; r < 8; ++r) {
    float v = acc[0][r];
#pragma unroll
    for (int j = 1; j < 8; ++j) v = fmaxf(v, acc[j][r]);
#pragma unroll
    for (int s = 1; s < 16; s <<= 1) v = fmaxf(v, __shfl_xor(v, s, 32));
    mx[r] = v;
  }
  if (nlane == 0) {
#pragma unroll
    for (int r = 0; r < 8; ++r) part[wave][mhalf + r] = mx[r];
  }
  __syncthreads();
  if (tid < 16) {
    float v = part[0][tid];
#pragma unroll
    for (int w = 1; w < 16; ++w) v = fmaxf(v, part[w][tid]);
    rowv[tid] = v;
  }
  __syncthreads();
  float rm[8];
#pragma unroll
  for (int r = 0; r < 8; ++r) rm[r] = rowv[mhalf + r];
  __syncthreads();

  // ---- exp + row sum ----
  float sm[8];
#pragma unroll
  for (int r = 0; r < 8; ++r) sm[r] = 0.0f;
#pragma unroll
  for (int j = 0; j < 8; ++j)
#pragma unroll
    for (int r = 0; r < 8; ++r) {
      float v = __expf(acc[j][r] - rm[r]);
      acc[j][r] = v;
      sm[r] += v;
    }
#pragma unroll
  for (int r = 0; r < 8; ++r) {
    float v = sm[r];
#pragma unroll
    for (int s = 1; s < 16; s <<= 1) v += __shfl_xor(v, s, 32);
    sm[r] = v;
  }
  if (nlane == 0) {
#pragma unroll
    for (int r = 0; r < 8; ++r) part[wave][mhalf + r] = sm[r];
  }
  __syncthreads();
  if (tid < 16) {
    float v = part[0][tid];
#pragma unroll
    for (int w = 1; w < 16; ++w) v += part[w][tid];
    rowv[tid] = 1.0f / v;
  }
  __syncthreads();
  float inv[8];
#pragma unroll
  for (int r = 0; r < 8; ++r) inv[r] = rowv[mhalf + r];
  __syncthreads();  // all rowv/part reads done before smem reuse below

  // ---- normalize + store aw (f32); stage bf16 P into wave-private LDS ----
  u16* psh = (u16*)smem + (size_t)wave * 2048;  // 16 rows x 128 cols bf16 = 4KB
#pragma unroll
  for (int j = 0; j < 8; ++j) {
    int n = n0 + 16 * j + nlane;
#pragma unroll
    for (int r = 0; r < 8; ++r) {
      int m = m0 + mhalf + r;
      float v = acc[j][r] * inv[r];
      e[(size_t)m * SKLEN + n] = v;
      psh[(mhalf + r) * 128 + 16 * j + nlane] = f32_bf16(v);
    }
  }
  // Same-wave LDS ops are in order; reads below see this wave's writes.

  // ---- P @ V over this wave's 128 K-columns -> partial 16x64 ----
  const u16* Vb = Vt + (size_t)bh * DK * SKLEN;
  v8f accO[4] = {};
#pragma unroll
  for (int kk = 0; kk < 128; kk += 32) {
    v16bf a = load_frag_a_lds(psh, kk, lane);
#pragma unroll
    for (int t = 0; t < 4; ++t) {
      v16bf bf = load_frag_b(Vb, SKLEN, 16 * t, n0 + kk, lane);
      accO[t] = __builtin_amdgcn_wmma_f32_16x16x32_bf16(
          false, a, false, bf, (short)0, accO[t], false, false);
    }
  }

  // ---- write per-wave partials (reusing this wave's own 4KB region) ----
  float* po = smem + (size_t)wave * 1024;  // 16 rows x 64 cols f32 = 4KB
#pragma unroll
  for (int t = 0; t < 4; ++t)
#pragma unroll
    for (int r = 0; r < 8; ++r)
      po[(mhalf + r) * 64 + 16 * t + nlane] = accO[t][r];
  __syncthreads();

  // ---- block reduction of 16 partials -> Ctx (bf16) ----
  u16* Cb = Ctx + (size_t)b * SQLEN * SADIM + h * DK;
#pragma unroll
  for (int ee = 0; ee < 2; ++ee) {
    int el = tid + ee * 512;           // 0..1023: m = el/64, d = el%64
    float s = smem[el];
#pragma unroll
    for (int w = 1; w < 16; ++w) s += smem[w * 1024 + el];
    int m = m0 + (el >> 6);
    int d = el & 63;
    Cb[(size_t)m * SADIM + d] = f32_bf16(s);
  }
}

// ---------------------------- host launch ----------------------------

extern "C" void kernel_launch(void* const* d_in, const int* in_sizes, int n_in,
                              void* d_out, int out_size, void* d_ws, size_t ws_size,
                              hipStream_t stream) {
  (void)in_sizes; (void)n_in; (void)out_size; (void)ws_size;
  const float* key_in   = (const float*)d_in[0];
  const float* value_in = (const float*)d_in[1];
  const float* query_in = (const float*)d_in[2];
  const float* Wk = (const float*)d_in[3];
  const float* bk = (const float*)d_in[4];
  const float* Wv = (const float*)d_in[5];
  const float* bv = (const float*)d_in[6];
  const float* Wq = (const float*)d_in[7];
  const float* bq = (const float*)d_in[8];
  const float* Wo = (const float*)d_in[9];
  const float* bo = (const float*)d_in[10];
  const int* mask = (const int*)d_in[11];

  float* out_cv = (float*)d_out;                                // [B, QLEN, KDIM]
  float* out_aw = out_cv + (size_t)BSZ * SQLEN * SKDIM;         // [B, H, QLEN, KLEN]

  char* ws = (char*)d_ws;
  size_t off = 0;
  auto wsAlloc = [&](size_t bytes) -> void* {
    void* p = ws + off;
    off += (bytes + 255) & ~(size_t)255;
    return p;
  };
  const size_t actBytes = (size_t)BSZ * SKLEN * SADIM * sizeof(u16);  // 4 MB
  const size_t wBytes   = (size_t)512 * 512 * sizeof(u16);            // 0.5 MB
  u16* Xk  = (u16*)wsAlloc(actBytes);
  u16* Xv  = (u16*)wsAlloc(actBytes);
  u16* Xq  = (u16*)wsAlloc(actBytes);
  u16* WkT = (u16*)wsAlloc(wBytes);
  u16* WvT = (u16*)wsAlloc(wBytes);
  u16* WqT = (u16*)wsAlloc(wBytes);
  u16* WoT = (u16*)wsAlloc(wBytes);
  u16* Kb  = (u16*)wsAlloc(actBytes);
  u16* Vb  = (u16*)wsAlloc(actBytes);
  u16* Qb  = (u16*)wsAlloc(actBytes);
  u16* Vt  = (u16*)wsAlloc((size_t)BSZ * NH * DK * SKLEN * sizeof(u16));
  u16* Ctx = (u16*)wsAlloc(actBytes);

  const int M = BSZ * SKLEN;  // 4096 rows for projection GEMMs
  const int n4 = BSZ * SKLEN * SKDIM / 4;

  // 1. convert inputs to bf16
  f32_to_bf16_vec<<<n4 / 256, 256, 0, stream>>>(key_in, Xk, n4);
  f32_to_bf16_vec<<<n4 / 256, 256, 0, stream>>>(value_in, Xv, n4);
  f32_to_bf16_vec<<<n4 / 256, 256, 0, stream>>>(query_in, Xq, n4);

  // 2. transpose + convert weights
  dim3 tb(32, 8);
  transpose512_to_bf16<<<dim3(16, 16), tb, 0, stream>>>(Wk, WkT);
  transpose512_to_bf16<<<dim3(16, 16), tb, 0, stream>>>(Wv, WvT);
  transpose512_to_bf16<<<dim3(16, 16), tb, 0, stream>>>(Wq, WqT);
  transpose512_to_bf16<<<dim3(16, 16), tb, 0, stream>>>(Wo, WoT);

  // 3. projections: K/V/Q = X @ W + b   (bf16 out)
  gemm_bf16_tn<0><<<dim3(M / 128, SADIM / 128), 256, 0, stream>>>(
      Xk, WkT, bk, Kb, nullptr, M, SADIM, SKDIM);
  gemm_bf16_tn<0><<<dim3(M / 128, SADIM / 128), 256, 0, stream>>>(
      Xv, WvT, bv, Vb, nullptr, M, SADIM, SKDIM);
  gemm_bf16_tn<0><<<dim3(M / 128, SADIM / 128), 256, 0, stream>>>(
      Xq, WqT, bq, Qb, nullptr, M, SADIM, SKDIM);

  // 4. per-head transpose of V -> [B*H][DK][KLEN]
  transpose_v_heads<<<dim3(SKLEN / 32, DK / 32, BSZ * NH), tb, 0, stream>>>(Vb, Vt);

  // 5. fused scores + mask + softmax + P@V -> aw (f32) and Ctx (bf16)
  attn_fused<<<dim3(SQLEN / 16, BSZ * NH), 512, 0, stream>>>(
      Qb, Kb, mask, out_aw, Vt, Ctx);

  // 6. output projection: cv = Ctx @ Wo + bo (f32 out)
  gemm_bf16_tn<1><<<dim3(M / 128, SKDIM / 128), 256, 0, stream>>>(
      Ctx, WoT, bo, nullptr, out_cv, M, SKDIM, SADIM);
}
